// S4Block_50766513439171
// MI455X (gfx1250) — matrix operand order repaired and checked by reading, not compiled, not run on hardware
//
#include <hip/hip_runtime.h>
#include <hip/hip_bf16.h>

typedef __attribute__((ext_vector_type(16))) _Float16     v16h;
typedef __attribute__((ext_vector_type(8)))  float        v8f;
typedef __attribute__((ext_vector_type(4)))  unsigned int u32x4;
typedef __attribute__((ext_vector_type(8)))  int          i32x8;
typedef __attribute__((ext_vector_type(4)))  int          i32x4;

#define D_MODEL 1024
#define SEQ_L   2048
#define BATCH   8
#define D_STATE 64
#define ROWS    (BATCH * SEQ_L)   // 16384

// ---------------------------------------------------------------------------
// Load 16 contiguous-ish f32 (two 32B chunks) and produce f16 hi/lo split
// fragments for WMMA (elements 0..7 from p0[0..7], 8..15 from p1[0..7]).
// ---------------------------------------------------------------------------
__device__ __forceinline__ void split16(const float* __restrict__ p0,
                                        const float* __restrict__ p1,
                                        v16h& hi, v16h& lo) {
  const float4* a = reinterpret_cast<const float4*>(p0);
  const float4* b = reinterpret_cast<const float4*>(p1);
  float4 x0 = a[0], x1 = a[1], x2 = b[0], x3 = b[1];
  float v[16] = {x0.x, x0.y, x0.z, x0.w, x1.x, x1.y, x1.z, x1.w,
                 x2.x, x2.y, x2.z, x2.w, x3.x, x3.y, x3.z, x3.w};
#pragma unroll
  for (int j = 0; j < 16; ++j) {
    _Float16 h = (_Float16)v[j];
    hi[j] = h;
    lo[j] = (_Float16)(v[j] - (float)h);
  }
}

// ---------------------------------------------------------------------------
// TDM: DMA one strided f32 column (SEQ_L rows, row stride D_MODEL elems)
// from global memory into contiguous LDS via tensor_load_to_lds.
// D# per CDNA5 ISA §8.3/8.4: 2-D tensor, tile = 1 x SEQ_L, data_size = 4B.
// Tracked by TENSORcnt; caller waits with s_wait_tensorcnt.
// ---------------------------------------------------------------------------
__device__ __forceinline__ void tdm_load_col(const float* gsrc, void* lds_dst) {
  const unsigned long long ga = (unsigned long long)(size_t)gsrc;
  const unsigned int laddr = (unsigned int)(size_t)lds_dst;  // LDS byte offset
  // Group 0: count=1 (valid user D#), lds_addr, global_addr[56:0], type=2
  u32x4 g0 = {1u,
              laddr,
              (unsigned int)ga,
              (unsigned int)((ga >> 32) & 0x01ffffffu) | (2u << 30)};
  // Group 1: data_size=2 (4B); tensor_dim0=D_MODEL; tensor_dim1=SEQ_L;
  //          tile_dim0=1; tile_dim1=SEQ_L; tensor_dim0_stride=D_MODEL.
  i32x8 g1 = {(int)(2u << 16),                               // data_size=4B
              (int)(((unsigned)D_MODEL & 0xffffu) << 16),    // tensor_dim0 lo16
              (int)((((unsigned)D_MODEL >> 16) & 0xffffu) |  // tensor_dim0 hi16
                    (((unsigned)SEQ_L & 0xffffu) << 16)),    // tensor_dim1 lo16
              (int)(1u << 16),                               // dim1 hi16=0 | tile_dim0=1
              (int)((unsigned)SEQ_L & 0xffffu),              // tile_dim1 | tile_dim2=0
              (int)(unsigned)D_MODEL,                        // tensor_dim0_stride lo32
              0, 0};                                         // stride hi / dim1_stride
  i32x4 z4 = {0, 0, 0, 0};
#if defined(__clang_major__) && (__clang_major__ >= 23)
  i32x8 z8 = {0, 0, 0, 0, 0, 0, 0, 0};
  __builtin_amdgcn_tensor_load_to_lds(g0, g1, z4, z4, z8, 0);
#else
  __builtin_amdgcn_tensor_load_to_lds(g0, g1, z4, z4, 0);
#endif
}

// ---------------------------------------------------------------------------
// Kernel 1: U = X @ W^T + b   (16384x1024)(1024x1024), f16x3-split WMMA.
// Block = 256 threads = 8 waves; block tile 128(M) x 64(N); wave tile 16x64.
// ---------------------------------------------------------------------------
__global__ __launch_bounds__(256) void k_inproj_gemm(
    const float* __restrict__ X, const float* __restrict__ W,
    const float* __restrict__ bias, float* __restrict__ U) {
  const int lane  = threadIdx.x & 31;
  const int wv    = threadIdx.x >> 5;
  const int m0    = blockIdx.y * 128 + wv * 16;
  const int n0    = blockIdx.x * 64;
  const int arow  = m0 + (lane & 15);
  const int abase = (lane < 16) ? 0 : 8;    // A: K split per ISA 16-bit A table
  const int bcol  = n0 + (lane & 15);
  const int bsel  = (lane < 16) ? 0 : 8;    // B: K split per ISA 16-bit table (same as A)

  v8f acc[4] = {};
  for (int k0 = 0; k0 < D_MODEL; k0 += 32) {
    v16h ahi, alo;
    const float* ap = X + (size_t)arow * D_MODEL + k0 + abase;
    __builtin_prefetch(ap + 32, 0, 1);      // global_prefetch_b8: next K chunk
    split16(ap, ap + 16, ahi, alo);
#pragma unroll
    for (int t = 0; t < 4; ++t) {
      v16h bhi, blo;
      const float* bp = W + (size_t)(bcol + 16 * t) * D_MODEL + k0 + bsel;
      split16(bp, bp + 16, bhi, blo);
      v8f c = acc[t];
      c = __builtin_amdgcn_wmma_f32_16x16x32_f16(false, ahi, false, bhi, (short)0, c, false, false);
      c = __builtin_amdgcn_wmma_f32_16x16x32_f16(false, ahi, false, blo, (short)0, c, false, false);
      c = __builtin_amdgcn_wmma_f32_16x16x32_f16(false, alo, false, bhi, (short)0, c, false, false);
      acc[t] = c;
    }
  }
#pragma unroll
  for (int t = 0; t < 4; ++t) {
    const int col = n0 + 16 * t + (lane & 15);
    const float bb = bias[col];
#pragma unroll
    for (int j = 0; j < 8; ++j) {
      const int r = m0 + ((lane < 16) ? j : (8 + j));   // C layout per ISA
      U[(size_t)r * D_MODEL + col] = acc[t][j] + bb;
    }
  }
}

// ---------------------------------------------------------------------------
// Kernel 2: in-place LayerNorm over rows of U (1024 elems/row).
// ---------------------------------------------------------------------------
__global__ __launch_bounds__(256) void k_layernorm(
    float* __restrict__ U, const float* __restrict__ gamma,
    const float* __restrict__ beta) {
  const int row = blockIdx.x;
  float* p = U + (size_t)row * D_MODEL;
  float v[4];
  float s = 0.f, s2 = 0.f;
#pragma unroll
  for (int i = 0; i < 4; ++i) {
    v[i] = p[threadIdx.x + 256 * i];
    s += v[i];
    s2 += v[i] * v[i];
  }
#pragma unroll
  for (int off = 16; off; off >>= 1) {
    s  += __shfl_xor(s,  off, 32);
    s2 += __shfl_xor(s2, off, 32);
  }
  __shared__ float sh[2][8];
  const int lane = threadIdx.x & 31, wv = threadIdx.x >> 5;
  if (lane == 0) { sh[0][wv] = s; sh[1][wv] = s2; }
  __syncthreads();
  s = 0.f; s2 = 0.f;
#pragma unroll
  for (int i = 0; i < 8; ++i) { s += sh[0][i]; s2 += sh[1][i]; }
  const float mu   = s * (1.f / 1024.f);
  const float var  = s2 * (1.f / 1024.f) - mu * mu;
  const float rstd = rsqrtf(var + 1e-5f);
#pragma unroll
  for (int i = 0; i < 4; ++i) {
    const int c = threadIdx.x + 256 * i;
    p[c] = (v[i] - mu) * rstd * gamma[c] + beta[c];
  }
}

// ---------------------------------------------------------------------------
// Kernel 3: P[l,n] = A^l * B  (complex);  store Re(P) and -Im(P).
// ---------------------------------------------------------------------------
__global__ __launch_bounds__(256) void k_gen_pows(
    const float* __restrict__ A_re, const float* __restrict__ A_im,
    const float* __restrict__ B_re, const float* __restrict__ B_im,
    float* __restrict__ Pre, float* __restrict__ Pim_neg) {
  const int idx = blockIdx.x * 256 + threadIdx.x;   // l*64 + n
  const int l = idx >> 6;
  const int n = idx & 63;
  const float ar = A_re[n], ai = A_im[n];
  const float lr = 0.5f * __logf(ar * ar + ai * ai);   // log|A|
  const float th = atan2f(ai, ar);                     // arg(A)
  const float er = __expf(lr * (float)l);
  float sn, cs;
  __sincosf(th * (float)l, &sn, &cs);
  const float pr = er * cs, pi = er * sn;              // A^l
  const float br = B_re[n], bi = B_im[n];
  Pre[idx]     =  pr * br - pi * bi;                   // Re(A^l * B)
  Pim_neg[idx] = -(pr * bi + pi * br);                 // -Im(A^l * B)
}

// ---------------------------------------------------------------------------
// Kernel 4: Kr[l,m] = Pre·C_re^T + (-Pim)·C_im^T + D_re[m]
// (2048 x 128) x (128 x 1024) GEMM, f16x3-split WMMA. Only the real part
// of K survives (x real, output takes .real), D_re on every tap per source.
// ---------------------------------------------------------------------------
__global__ __launch_bounds__(256) void k_kr_gemm(
    const float* __restrict__ Pre, const float* __restrict__ Pim_neg,
    const float* __restrict__ Cre, const float* __restrict__ Cim,
    const float* __restrict__ Dre, float* __restrict__ Kr) {
  const int lane  = threadIdx.x & 31;
  const int wv    = threadIdx.x >> 5;
  const int m0    = blockIdx.y * 128 + wv * 16;   // l tile
  const int n0    = blockIdx.x * 64;              // channel tile
  const int arow  = m0 + (lane & 15);
  const int abase = (lane < 16) ? 0 : 8;
  const int bcol  = n0 + (lane & 15);
  const int bsel  = (lane < 16) ? 0 : 8;

  v8f acc[4] = {};
#pragma unroll
  for (int k0 = 0; k0 < 128; k0 += 32) {
    const float* Asrc = (k0 < 64) ? Pre : Pim_neg;
    const float* Bsrc = (k0 < 64) ? Cre : Cim;
    const int kk = k0 & 63;
    v16h ahi, alo;
    const float* ap = Asrc + (size_t)arow * D_STATE + kk + abase;
    split16(ap, ap + 16, ahi, alo);
#pragma unroll
    for (int t = 0; t < 4; ++t) {
      v16h bhi, blo;
      const float* bp = Bsrc + (size_t)(bcol + 16 * t) * D_STATE + kk + bsel;
      split16(bp, bp + 16, bhi, blo);
      v8f c = acc[t];
      c = __builtin_amdgcn_wmma_f32_16x16x32_f16(false, ahi, false, bhi, (short)0, c, false, false);
      c = __builtin_amdgcn_wmma_f32_16x16x32_f16(false, ahi, false, blo, (short)0, c, false, false);
      c = __builtin_amdgcn_wmma_f32_16x16x32_f16(false, alo, false, bhi, (short)0, c, false, false);
      acc[t] = c;
    }
  }
#pragma unroll
  for (int t = 0; t < 4; ++t) {
    const int col = n0 + 16 * t + (lane & 15);
    const float dd = Dre[col];
#pragma unroll
    for (int j = 0; j < 8; ++j) {
      const int r = m0 + ((lane < 16) ? j : (8 + j));
      Kr[(size_t)r * D_MODEL + col] = acc[t][j] + dd;
    }
  }
}

// ---------------------------------------------------------------------------
// Kernel 5: causal depthwise Toeplitz conv + residual, per channel in LDS.
// One block per (channel m, batch-group of 4). 40 KB LDS.
// Column loads done by the Tensor Data Mover (tensor_load_to_lds), issued
// once by wave 0, synchronized with s_wait_tensorcnt + workgroup barrier.
//   y[b,t] = sum_{s<=t} Kr[s,m] * x2[b,t-s,m];  out = y + x2.
// ---------------------------------------------------------------------------
__global__ __launch_bounds__(256) void k_conv_residual(
    const float* __restrict__ U, const float* __restrict__ Kr,
    float* __restrict__ Y) {
  const int m  = blockIdx.x;   // channel 0..1023
  const int bg = blockIdx.y;   // batch group 0..1

  __shared__ float kcol[SEQ_L];
  __shared__ float xcol[4][SEQ_L];

  if (threadIdx.x < 32) {      // wave 0 issues the 5 TDM descriptors
    tdm_load_col(Kr + m, (void*)&kcol[0]);
#pragma unroll
    for (int b = 0; b < 4; ++b)
      tdm_load_col(U + ((size_t)(bg * 4 + b) * SEQ_L) * D_MODEL + m,
                   (void*)&xcol[b][0]);
    __builtin_amdgcn_s_wait_tensorcnt(0);
  }
  __syncthreads();

#pragma unroll
  for (int b = 0; b < 4; ++b) {
    float* dst = Y + ((size_t)(bg * 4 + b) * SEQ_L) * D_MODEL + m;
    for (int t = threadIdx.x; t < SEQ_L; t += 256) {
      float acc = 0.f;
      int s = 0;
      for (; s + 3 <= t; s += 4) {
        acc += kcol[s]     * xcol[b][t - s];
        acc += kcol[s + 1] * xcol[b][t - s - 1];
        acc += kcol[s + 2] * xcol[b][t - s - 2];
        acc += kcol[s + 3] * xcol[b][t - s - 3];
      }
      for (; s <= t; ++s) acc += kcol[s] * xcol[b][t - s];
      dst[(size_t)t * D_MODEL] = acc + xcol[b][t];   // residual
    }
  }
}

// ---------------------------------------------------------------------------
extern "C" void kernel_launch(void* const* d_in, const int* in_sizes, int n_in,
                              void* d_out, int out_size, void* d_ws, size_t ws_size,
                              hipStream_t stream) {
  (void)in_sizes; (void)n_in; (void)out_size; (void)ws_size;
  const float* x     = (const float*)d_in[0];
  const float* W_in  = (const float*)d_in[1];
  const float* b_in  = (const float*)d_in[2];
  const float* gamma = (const float*)d_in[3];
  const float* beta  = (const float*)d_in[4];
  const float* A_re  = (const float*)d_in[5];
  const float* A_im  = (const float*)d_in[6];
  const float* B_re  = (const float*)d_in[7];
  const float* B_im  = (const float*)d_in[8];
  const float* C_re  = (const float*)d_in[9];
  const float* C_im  = (const float*)d_in[10];
  const float* D_re  = (const float*)d_in[11];
  // d_in[12] = D_im is unused: conv input is real and only Re(y) is kept.
  float* out = (float*)d_out;

  float* U   = (float*)d_ws;                          // 16384 x 1024 f32
  float* Pre = U + (size_t)ROWS * D_MODEL;            // 2048 x 64
  float* Pim = Pre + (size_t)SEQ_L * D_STATE;         // 2048 x 64
  float* Kr  = Pim + (size_t)SEQ_L * D_STATE;         // 2048 x 1024

  k_inproj_gemm<<<dim3(D_MODEL / 64, ROWS / 128), 256, 0, stream>>>(x, W_in, b_in, U);
  k_layernorm<<<ROWS, 256, 0, stream>>>(U, gamma, beta);
  k_gen_pows<<<(SEQ_L * D_STATE) / 256, 256, 0, stream>>>(A_re, A_im, B_re, B_im, Pre, Pim);
  k_kr_gemm<<<dim3(D_MODEL / 64, SEQ_L / 128), 256, 0, stream>>>(Pre, Pim, C_re, C_im, D_re, Kr);
  k_conv_residual<<<dim3(D_MODEL, 2), 256, 0, stream>>>(U, Kr, out);
}
